// PSLAGFU_29609504538952
// MI455X (gfx1250) — matrix operand order repaired and checked by reading, not compiled, not run on hardware
//
#include <hip/hip_runtime.h>
#include <hip/hip_bf16.h>
#include <math.h>

// MI455X (gfx1250) implementation.
// Compute-bound workload (~1.6 TFLOP of 3x3 convs vs ~0.6GB traffic), so all
// three conv layers per level are implicit-GEMM on v_wmma_f32_16x16x32_f16
// (f16 inputs, f32 accumulate). wx/wh convs are fused into one GEMM over the
// concatenated [gating ; h_pre] input channels. Weight tiles are staged into
// LDS with gfx1250 async-to-LDS copies (ASYNCcnt), K is unrolled 64-wide per
// barrier step, and im2col index math is an incremental (ci,ty,tx) walk.

typedef __attribute__((ext_vector_type(16))) _Float16 v16h;
typedef __attribute__((ext_vector_type(8)))  _Float16 v8h;
typedef __attribute__((ext_vector_type(8)))  float    v8f;

union Frag16 { v16h v; v8h h2[2]; _Float16 h[16]; };
union H8     { v8h v; _Float16 h[8]; };

__device__ __forceinline__ float sigmoidf_(float v) { return 1.0f / (1.0f + __expf(-v)); }

// ---------------------------------------------------------------------------
// 1) Local 3x3 correlation + softmax over offsets + weighted aggregation.
//    One 256-thread block per output pixel; block-reduce over C channels.
// ---------------------------------------------------------------------------
__global__ void corr_softmax_agg(const float* __restrict__ xp,  // feat1 [B,C,S,S]
                                 const float* __restrict__ xc,  // feat2 [B,C,S,S]
                                 float* __restrict__ nf,        // new_feat out
                                 int C, int S)
{
    const int pix = blockIdx.x;                 // b*S*S + y*S + x
    const int SS  = S * S;
    const int b   = pix / SS;
    const int rem = pix - b * SS;
    const int y   = rem / S;
    const int x   = rem - y * S;
    const int tid = threadIdx.x;

    __shared__ float red[9 * 256];
    float corr9[9];
    #pragma unroll
    for (int k = 0; k < 9; ++k) corr9[k] = 0.f;

    const size_t basebc = (size_t)b * C * SS;
    for (int c = tid; c < C; c += 256) {
        const float f2 = xc[basebc + (size_t)c * SS + rem];
        #pragma unroll
        for (int k = 0; k < 9; ++k) {
            const int yy = y + k / 3 - 1;
            const int xx = x + (k % 3) - 1;
            float p = 0.f;
            if (yy >= 0 && yy < S && xx >= 0 && xx < S)
                p = xp[basebc + (size_t)c * SS + yy * S + xx];
            corr9[k] += f2 * p;
        }
    }
    #pragma unroll
    for (int k = 0; k < 9; ++k) red[k * 256 + tid] = corr9[k];
    __syncthreads();
    for (int s = 128; s > 0; s >>= 1) {
        if (tid < s) {
            #pragma unroll
            for (int k = 0; k < 9; ++k) red[k * 256 + tid] += red[k * 256 + tid + s];
        }
        __syncthreads();
    }
    // softmax over the 9 offsets (redundant per-thread, registers only)
    float attn[9];
    float mx = red[0];
    #pragma unroll
    for (int k = 1; k < 9; ++k) mx = fmaxf(mx, red[k * 256]);
    float sum = 0.f;
    #pragma unroll
    for (int k = 0; k < 9; ++k) { attn[k] = __expf(red[k * 256] - mx); sum += attn[k]; }
    const float inv = 1.f / sum;
    #pragma unroll
    for (int k = 0; k < 9; ++k) attn[k] *= inv;

    for (int c = tid; c < C; c += 256) {
        float acc = 0.f;
        #pragma unroll
        for (int k = 0; k < 9; ++k) {
            const int yy = y + k / 3 - 1;
            const int xx = x + (k % 3) - 1;
            if (yy >= 0 && yy < S && xx >= 0 && xx < S)
                acc += attn[k] * xp[basebc + (size_t)c * SS + yy * S + xx];
        }
        nf[basebc + (size_t)c * SS + rem] = acc;
    }
}

// ---------------------------------------------------------------------------
// 2) 1-output-channel 3x3 convs (wp & wq) over feat_cat=[new_feat, feat2],
//    fused, sigmoid epilogue. One block per pixel, reduce over 2C*9 terms.
// ---------------------------------------------------------------------------
__global__ void pw_gate_conv(const float* __restrict__ nf, const float* __restrict__ xc,
                             const float* __restrict__ wp, const float* __restrict__ bp,
                             const float* __restrict__ wq, const float* __restrict__ bq,
                             float* __restrict__ prevw, float* __restrict__ presw,
                             int C, int S)
{
    const int pix = blockIdx.x;
    const int SS  = S * S;
    const int b   = pix / SS;
    const int rem = pix - b * SS;
    const int y   = rem / S;
    const int x   = rem - y * S;
    const int tid = threadIdx.x;

    __shared__ float rp[256], rq[256];
    float ap = 0.f, aq = 0.f;
    const size_t basebc = (size_t)b * C * SS;
    for (int ci = tid; ci < 2 * C; ci += 256) {
        const float* src = (ci < C) ? nf : xc;
        const int c = (ci < C) ? ci : ci - C;
        #pragma unroll
        for (int t = 0; t < 9; ++t) {
            const int yy = y + t / 3 - 1;
            const int xx = x + (t % 3) - 1;
            if (yy < 0 || yy >= S || xx < 0 || xx >= S) continue;
            const float v = src[basebc + (size_t)c * SS + yy * S + xx];
            ap += wp[ci * 9 + t] * v;
            aq += wq[ci * 9 + t] * v;
        }
    }
    rp[tid] = ap; rq[tid] = aq;
    __syncthreads();
    for (int s = 128; s > 0; s >>= 1) {
        if (tid < s) { rp[tid] += rp[tid + s]; rq[tid] += rq[tid + s]; }
        __syncthreads();
    }
    if (tid == 0) {
        prevw[pix] = sigmoidf_(rp[0] + bp[0]);
        presw[pix] = sigmoidf_(rq[0] + bq[0]);
    }
}

// ---------------------------------------------------------------------------
// 3) gating_in (f16) = [new_feat*prev_w ; feat2*pres_w]   [B,2C,S,S]
// ---------------------------------------------------------------------------
__global__ void build_gating_in(const float* __restrict__ nf, const float* __restrict__ xc,
                                const float* __restrict__ prevw, const float* __restrict__ presw,
                                _Float16* __restrict__ out, int C, int S, int total)
{
    const int idx = blockIdx.x * 256 + threadIdx.x;
    if (idx >= total) return;
    const int SS = S * S;
    int t = idx;
    const int pos = t % SS; t /= SS;
    const int cc  = t % (2 * C); t /= (2 * C);
    const int b   = t;
    float v;
    if (cc < C) v = nf[((size_t)b * C + cc) * SS + pos] * prevw[b * SS + pos];
    else        v = xc[((size_t)b * C + (cc - C)) * SS + pos] * presw[b * SS + pos];
    out[idx] = (_Float16)v;
}

// ---------------------------------------------------------------------------
// Weight casts to f16 GEMM-A layout [Cout][Cin*9]
// ---------------------------------------------------------------------------
__global__ void cvt_f32_f16(const float* __restrict__ src, _Float16* __restrict__ dst, int n)
{
    const int i = blockIdx.x * 256 + threadIdx.x;
    if (i < n) dst[i] = (_Float16)src[i];
}

// interleave wx [4C][C][9] and wh [4C][C][9] -> A2 [4C][2C*9]
__global__ void cvt_wxh(const float* __restrict__ wx, const float* __restrict__ wh,
                        _Float16* __restrict__ dst, int C, int n)
{
    const int i = blockIdx.x * 256 + threadIdx.x;
    if (i >= n) return;
    const int t  = i % 9;
    const int r  = i / 9;
    const int ci = r % (2 * C);
    const int o  = r / (2 * C);
    const float v = (ci < C) ? wx[((size_t)o * C + ci) * 9 + t]
                             : wh[((size_t)o * C + (ci - C)) * 9 + t];
    dst[i] = (_Float16)v;
}

// h_pre f32 -> f16 into second half (channels C..2C-1) of the xh buffer
__global__ void pack_h_f16(const float* __restrict__ h, _Float16* __restrict__ xh,
                           int C, int S, int total)
{
    const int idx = blockIdx.x * 256 + threadIdx.x;
    if (idx >= total) return;
    const int SS = S * S;
    int t = idx;
    const int pos = t % SS; t /= SS;
    const int c   = t % C;  t /= C;
    const int b   = t;
    xh[((size_t)b * 2 * C + C + c) * SS + pos] = (_Float16)h[idx];
}

// ---------------------------------------------------------------------------
// 4) Implicit-GEMM 3x3 conv with v_wmma_f32_16x16x32_f16.
//    D[M=Cout][N=B*H*W] = A[M][K=Cin*9] x im2col(B)[K][N]
//    Tile: 128(M) x 64(N) per 256-thread block (8 wave32, each 32x32 = 2x2
//    frags). K stepped 64-wide per barrier (2 WMMA sub-steps). A tile staged
//    via gfx1250 GLOBAL_LOAD_ASYNC_TO_LDS_B128 (ASYNCcnt); B tile im2col is
//    register-built (needs zero padding) with incremental (ci,ty,tx) walk.
//    M and K are always multiples of 128/64 for this network; N is guarded.
// ---------------------------------------------------------------------------
#define MT  128
#define NT  64
#define KS  64
#define LDA 72   // 64 + 8 halves pad (144B rows keep 16B alignment, dodge conflicts)
#define LDB 72

__global__ __launch_bounds__(256) void conv_gemm_wmma(
    const _Float16* __restrict__ A,   // [Cout][Cin*9] f16
    const _Float16* __restrict__ Bs,  // activations [B][Cin][H][W] f16
    int Cin, int H, int W, int N, int Cout,
    const float* __restrict__ bias1, const float* __restrict__ bias2,
    float* __restrict__ outF, int relu,
    _Float16* __restrict__ outH, int C16)
{
    __shared__ __align__(16) _Float16 lA[MT * LDA];
    __shared__ __align__(16) _Float16 lB[NT * LDB];

    const int tid  = threadIdx.x;
    const int lane = tid & 31;
    const int wv   = tid >> 5;       // 8 waves
    const int mw   = wv & 3;         // 4 waves along M (32 rows each)
    const int nw   = wv >> 2;        // 2 waves along N (32 cols each)

    const int n0 = blockIdx.x * NT;
    const int m0 = blockIdx.y * MT;
    const int K  = Cin * 9;          // always a multiple of 64 here
    const int HW = H * W;

    v8f zero = {};
    v8f acc[2][2];
    acc[0][0] = zero; acc[0][1] = zero; acc[1][0] = zero; acc[1][1] = zero;

    // A staging map: 128 rows x 64 cols, 32 halves (4 x b128) per thread
    const int arow = tid >> 1;
    const int acol = (tid & 1) << 5;
    const unsigned ldsA = (unsigned)(size_t)(lA + arow * LDA + acol); // flat->LDS low bits
    // B staging map: 64 n-rows x 4 k-chunks of 16
    const int bnn = tid >> 2;
    const int bkc = (tid & 3) << 4;
    const int ngl = n0 + bnn;
    int bb = 0, yy0 = 0, xx0 = 0;
    if (ngl < N) {
        bb = ngl / HW;
        const int r = ngl - bb * HW;
        yy0 = r / W;
        xx0 = r - yy0 * W;
    }

    for (int k0 = 0; k0 < K; k0 += KS) {
        // ---- stage A (weights): async global->LDS, tracked by ASYNCcnt ----
        {
            const _Float16* src = A + (size_t)(m0 + arow) * K + (k0 + acol);
            #pragma unroll
            for (int u = 0; u < 4; ++u) {
                asm volatile("global_load_async_to_lds_b128 %0, %1, off"
                             :
                             : "v"(ldsA + 16u * u),
                               "v"((unsigned long long)(size_t)(src + 8 * u))
                             : "memory");
            }
            if (k0 + KS < K) __builtin_prefetch(src + KS, 0, 1);   // global_prefetch_b8
        }
        // ---- stage B (im2col, transposed: lB[n][k]), incremental index walk ----
        {
            int k  = k0 + bkc;
            int ci = k / 9;
            int tt = k - ci * 9;
            int ty = tt / 3;
            int tx = tt - ty * 3;
            H8 t0, t1;
            #pragma unroll
            for (int j = 0; j < 16; ++j) {
                _Float16 v = (_Float16)0.f;
                if (ngl < N) {
                    const int yy = yy0 + ty - 1;
                    const int xx = xx0 + tx - 1;
                    if (yy >= 0 && yy < H && xx >= 0 && xx < W)
                        v = Bs[((size_t)bb * Cin + ci) * HW + yy * W + xx];
                }
                if (j < 8) t0.h[j] = v; else t1.h[j - 8] = v;
                if (++tx == 3) { tx = 0; if (++ty == 3) { ty = 0; ++ci; } }
            }
            *(v8h*)(lB + bnn * LDB + bkc)     = t0.v;
            *(v8h*)(lB + bnn * LDB + bkc + 8) = t1.v;
        }
        asm volatile("s_wait_asynccnt 0x0" ::: "memory");
        __syncthreads();

        // ---- fragments + WMMA: two K=32 sub-steps per barrier ----
        #pragma unroll
        for (int ks = 0; ks < KS; ks += 32) {
            Frag16 af[2], bf[2];
            const int ml = lane & 15;
            const int ka = ks + ((lane >> 4) << 3);   // A: +0 or +8
            const int kb = ks + ((lane >> 4) << 3);   // B: +0 or +8
            #pragma unroll
            for (int i = 0; i < 2; ++i) {
                const int m = mw * 32 + i * 16 + ml;
                af[i].h2[0] = *(const v8h*)(lA + m * LDA + ka);
                af[i].h2[1] = *(const v8h*)(lA + m * LDA + ka + 16);
            }
            #pragma unroll
            for (int j = 0; j < 2; ++j) {
                const int nn = nw * 32 + j * 16 + ml;
                bf[j].h2[0] = *(const v8h*)(lB + nn * LDB + kb);
                bf[j].h2[1] = *(const v8h*)(lB + nn * LDB + kb + 16);
            }
            #pragma unroll
            for (int i = 0; i < 2; ++i)
                #pragma unroll
                for (int j = 0; j < 2; ++j)
                    acc[i][j] = __builtin_amdgcn_wmma_f32_16x16x32_f16(
                        false, af[i].v, false, bf[j].v, (short)0, acc[i][j], false, false);
        }
        __syncthreads();
    }

    // ---- epilogue: bias (+bias2), optional relu, f32 store (+ optional f16 copy)
    const int ml = lane & 15;
    const int mh = (lane >> 4) << 3;
    #pragma unroll
    for (int j = 0; j < 2; ++j) {
        const int n = n0 + nw * 32 + j * 16 + ml;
        if (n >= N) continue;
        const int b = n / HW;
        const int r = n - b * HW;
        const int y = r / W;
        const int x = r - y * W;
        #pragma unroll
        for (int i = 0; i < 2; ++i) {
            #pragma unroll
            for (int v = 0; v < 8; ++v) {
                const int m = m0 + mw * 32 + i * 16 + mh + v;
                float val = acc[i][j][v] + bias1[m] + (bias2 ? bias2[m] : 0.f);
                if (relu) val = fmaxf(val, 0.f);
                outF[((size_t)b * Cout + m) * HW + y * W + x] = val;
                if (outH) outH[((size_t)b * C16 + m) * HW + y * W + x] = (_Float16)val;
            }
        }
    }
}

// ---------------------------------------------------------------------------
// 5) ConvLSTM pointwise: gates from cx [B,4C,S,S], update h/c.
// ---------------------------------------------------------------------------
__global__ void lstm_pointwise(const float* __restrict__ cx, const float* __restrict__ cpre,
                               float* __restrict__ hout, float* __restrict__ cout,
                               int C, int S, int total)
{
    const int idx = blockIdx.x * 256 + threadIdx.x;
    if (idx >= total) return;
    const int SS = S * S;
    int t = idx;
    const int pos = t % SS; t /= SS;
    const int c   = t % C;  t /= C;
    const int b   = t;
    const size_t gstride = (size_t)C * SS;
    const size_t base    = ((size_t)b * 4 * C + c) * SS + pos;
    const float ig = sigmoidf_(cx[base]);
    const float fg = sigmoidf_(cx[base + gstride]);
    const float og = sigmoidf_(cx[base + 2 * gstride]);
    const float gg = fmaxf(cx[base + 3 * gstride], 0.f);
    const float cn = fg * cpre[idx] + ig * gg;
    cout[idx] = cn;
    hout[idx] = og * fmaxf(cn, 0.f);
}

// ---------------------------------------------------------------------------
static const int CH[6] = {512, 1024, 512, 256, 256, 256};
static const int SZ[6] = {64, 32, 16, 8, 4, 2};
#define BN 4

extern "C" void kernel_launch(void* const* d_in, const int* in_sizes, int n_in,
                              void* d_out, int out_size, void* d_ws, size_t ws_size,
                              hipStream_t stream)
{
    (void)in_sizes; (void)n_in; (void)out_size; (void)ws_size;

    // output offsets: per level (h_next, c_next, gating), each B*C*S*S floats
    size_t off[6]; size_t a = 0;
    for (int l = 0; l < 6; ++l) { off[l] = a; a += 3ull * BN * CH[l] * SZ[l] * SZ[l]; }

    char* wsBase = (char*)d_ws;

    for (int l = 0; l < 6; ++l) {
        const int C = CH[l], S = SZ[l], SS = S * S;
        const int N = BN * SS;

        const float* xp = (const float*)d_in[4 * l + 0];
        const float* xc = (const float*)d_in[4 * l + 1];
        const float* hp = (const float*)d_in[4 * l + 2];
        const float* cp = (const float*)d_in[4 * l + 3];
        const float* wp = (const float*)d_in[24 + l];
        const float* bp = (const float*)d_in[30 + l];
        const float* wq = (const float*)d_in[36 + l];
        const float* bq = (const float*)d_in[42 + l];
        const float* wc = (const float*)d_in[48 + l];
        const float* bc = (const float*)d_in[54 + l];
        const float* wx = (const float*)d_in[60 + l];
        const float* bx = (const float*)d_in[66 + l];
        const float* wh = (const float*)d_in[72 + l];
        const float* bh = (const float*)d_in[78 + l];

        float* h_out = (float*)d_out + off[l];
        float* c_out = h_out + (size_t)BN * C * SS;
        float* g_out = c_out + (size_t)BN * C * SS;

        // bump allocator over d_ws; regions reused across levels (stream-ordered)
        size_t cur = 0;
        auto alloc = [&](size_t bytes) -> char* {
            cur = (cur + 255) & ~(size_t)255;
            char* r = wsBase + cur;
            cur += bytes;
            return r;
        };
        _Float16* A1  = (_Float16*)alloc((size_t)C * 2 * C * 9 * sizeof(_Float16));
        _Float16* A2  = (_Float16*)alloc((size_t)4 * C * 2 * C * 9 * sizeof(_Float16));
        float*    nf  = (float*)   alloc((size_t)BN * C * SS * sizeof(float));
        float*    pvw = (float*)   alloc((size_t)BN * SS * sizeof(float));
        float*    psw = (float*)   alloc((size_t)BN * SS * sizeof(float));
        _Float16* gin = (_Float16*)alloc((size_t)BN * 2 * C * SS * sizeof(_Float16));
        _Float16* xh  = (_Float16*)alloc((size_t)BN * 2 * C * SS * sizeof(_Float16));
        float*    cx  = (float*)   alloc((size_t)BN * 4 * C * SS * sizeof(float));

        { const int n = C * 2 * C * 9;
          cvt_f32_f16<<<(n + 255) / 256, 256, 0, stream>>>(wc, A1, n); }
        { const int n = 4 * C * 2 * C * 9;
          cvt_wxh<<<(n + 255) / 256, 256, 0, stream>>>(wx, wh, A2, C, n); }

        corr_softmax_agg<<<N, 256, 0, stream>>>(xp, xc, nf, C, S);
        pw_gate_conv<<<N, 256, 0, stream>>>(nf, xc, wp, bp, wq, bq, pvw, psw, C, S);
        { const int n = BN * 2 * C * SS;
          build_gating_in<<<(n + 255) / 256, 256, 0, stream>>>(nf, xc, pvw, psw, gin, C, S, n); }

        // gating = relu(conv(wc, gating_in)); writes f32 output + f16 half of xh
        { dim3 g((N + NT - 1) / NT, C / MT);
          conv_gemm_wmma<<<g, 256, 0, stream>>>(A1, gin, 2 * C, S, S, N, C,
                                                bc, nullptr, g_out, 1, xh, 2 * C); }
        { const int n = BN * C * SS;
          pack_h_f16<<<(n + 255) / 256, 256, 0, stream>>>(hp, xh, C, S, n); }

        // cx = conv(wx, gating) + bx + conv(wh, h_pre) + bh  (fused single GEMM)
        { dim3 g((N + NT - 1) / NT, (4 * C) / MT);
          conv_gemm_wmma<<<g, 256, 0, stream>>>(A2, xh, 2 * C, S, S, N, 4 * C,
                                                bx, bh, cx, 0, nullptr, 0); }

        { const int n = BN * C * SS;
          lstm_pointwise<<<(n + 255) / 256, 256, 0, stream>>>(cx, cp, h_out, c_out, C, S, n); }
    }
}
